// STCCapsNet_17308718203517
// MI455X (gfx1250) — matrix operand order repaired and checked by reading, compiled, not run on hardware
//
#include <hip/hip_runtime.h>
#include <math.h>

typedef __attribute__((ext_vector_type(16))) _Float16 v16h;
typedef __attribute__((ext_vector_type(8)))  _Float16 v8h;
typedef __attribute__((ext_vector_type(8)))  float    v8f;

#define BATCH 32
#define LDSS 40   // padded LDS row stride (halves): 80B = 20 banks -> conflict-free

static __device__ __forceinline__ v8h h8_zero() {
    v8h z;
    #pragma unroll
    for (int q = 0; q < 8; ++q) z[q] = (_Float16)0.0f;
    return z;
}

// ---------------------------------------------------------------------------
// Implicit-GEMM convolution on V_WMMA_F32_16X16X32_F16.
//   Activations NHWC f16; weights wT[oc][cell][ic] f16 zero-padded to KP.
//   GEMM: M = B*OH*OW (= NHWC pixel index), N = OC, k = cell*IC + ic.
//   Block: 128 threads = 4 wave32s; tile 64M x 64N; wave owns 32x32 ->
//   4 wmma per 32-deep K step. Double-buffered LDS: one barrier per K step,
//   next-tile global loads issue before current-tile wmma.
//   Epilogue geometry (OCT/CHOFF/RELU) is compile-time.
// ---------------------------------------------------------------------------
template<int OC, int IC, int IH, int IW, int KH, int KW,
         int SH, int SW, int PH, int PW, int OH, int OW,
         int OCT, int CHOFF, int RELU>
__global__ void __launch_bounds__(128) wmma_conv_kernel(
    const _Float16* __restrict__ in, const _Float16* __restrict__ wgt,
    const float* __restrict__ bias, _Float16* __restrict__ out)
{
    constexpr int K    = IC * KH * KW;
    constexpr int KP   = (K + 31) & ~31;
    constexpr int OHW  = OH * OW;
    constexpr int IHW  = IH * IW;
    constexpr bool FASTA   = (IC % 16) == 0;
    constexpr bool NEEDCHK = (PH != 0) || (PW != 0) || ((K % 32) != 0);

    __shared__ _Float16 As[2][64 * LDSS];
    __shared__ _Float16 Bs[2][64 * LDSS];

    const int tid  = threadIdx.x;
    const int wave = tid >> 5;
    const int lane = tid & 31;
    const int g    = lane >> 4;
    const int hl   = lane & 15;
    const int wm   = wave >> 1;
    const int wn   = wave & 1;

    const int m0     = blockIdx.y * 64;
    const int nBlock = blockIdx.x * 64;

    // loader: one LDS row per thread pair; 16 contiguous k each
    const int lrow = tid >> 1;
    const int kkb  = (tid & 1) * 16;

    // hoisted m-decode (constexpr divisors)
    const int am  = m0 + lrow;
    const int ab  = am / OHW;
    const int ar  = am - ab * OHW;
    const int aoh = ar / OW;
    const int aow = ar - aoh * OW;
    const int ihb = aoh * SH - PH;
    const int iwb = aow * SW - PW;
    const _Float16* __restrict__ aPix = in + (long)ab * IHW * IC;
    const _Float16* __restrict__ bRow = wgt + (long)(nBlock + lrow) * KP;

    auto stageA = [&](int buf, int k0) {
        const int kbase = k0 + kkb;
        v8h lo, hi;
        if (FASTA) {
            const int cell = kbase / IC;          // constexpr divisor
            const int ic0  = kbase - cell * IC;
            const int kh   = cell / KW;
            const int kw   = cell - kh * KW;
            int ih = ihb + kh;
            int iw = iwb + kw;
            bool inb = true;
            if (NEEDCHK) {
                inb = (ih >= 0) & (ih < IH) & (iw >= 0) & (iw < IW);
                if (!inb) { ih = 0; iw = 0; }
            }
            const _Float16* src = aPix + (long)(ih * IW + iw) * IC + ic0;
            lo = *(const v8h*)src;
            hi = *(const v8h*)(src + 8);
            if (NEEDCHK && !inb) { lo = h8_zero(); hi = h8_zero(); }
        } else {
            // elementwise gather (IC % 16 != 0: s1 IC==1, fusion IC==260)
            _Float16 tmp[16];
            #pragma unroll
            for (int j = 0; j < 16; ++j) {
                const int k    = kbase + j;
                const int cell = k / IC;
                const int ic   = k - cell * IC;
                const int kh   = cell / KW;
                const int kw   = cell - kh * KW;
                const int ih   = ihb + kh;
                const int iw   = iwb + kw;
                const bool inb = (ih >= 0) & (ih < IH) & (iw >= 0) & (iw < IW);
                const long idx = inb ? ((long)(ih * IW + iw) * IC + ic) : 0;
                const _Float16 v = aPix[idx];
                tmp[j] = inb ? v : (_Float16)0.0f;
            }
            #pragma unroll
            for (int j = 0; j < 8; ++j) { lo[j] = tmp[j]; hi[j] = tmp[8 + j]; }
        }
        v8h* dst = (v8h*)&As[buf][lrow * LDSS + kkb];
        dst[0] = lo; dst[1] = hi;
    };
    auto stageB = [&](int buf, int k0) {
        const _Float16* src = bRow + k0 + kkb;
        v8h lo = *(const v8h*)src;
        v8h hi = *(const v8h*)(src + 8);
        v8h* dst = (v8h*)&Bs[buf][lrow * LDSS + kkb];
        dst[0] = lo; dst[1] = hi;
    };

    v8f acc[2][2];
    #pragma unroll
    for (int u = 0; u < 2; ++u) {
        const float bv = bias[nBlock + wn * 32 + u * 16 + hl];
        #pragma unroll
        for (int t = 0; t < 2; ++t)
            #pragma unroll
            for (int r = 0; r < 8; ++r) acc[t][u][r] = bv;
    }

    stageA(0, 0);
    stageB(0, 0);
    __syncthreads();

    int cur = 0;
    for (int k0 = 0; k0 < KP; k0 += 32) {
        // ---- fragments from current buffer (ISA 7.12.2 layouts)
        v16h af[2], bf[2];
        #pragma unroll
        for (int t = 0; t < 2; ++t) {
            const int arow = (wm * 32 + t * 16 + hl) * LDSS;
            const int brow = (wn * 32 + t * 16 + hl) * LDSS;
            #pragma unroll
            for (int j = 0; j < 8; ++j) {
                const int kbA = ((j < 4) ? 0 : 16) + g * 8 + 2 * (j & 3);
                af[t][2 * j]     = As[cur][arow + kbA];
                af[t][2 * j + 1] = As[cur][arow + kbA + 1];
                const int kbB = ((j < 4) ? 0 : 16) + g * 8 + 2 * (j & 3);
                bf[t][2 * j]     = Bs[cur][brow + kbB];
                bf[t][2 * j + 1] = Bs[cur][brow + kbB + 1];
            }
        }
        // ---- prefetch/stage next tile into the other buffer
        if (k0 + 32 < KP) {
            stageA(cur ^ 1, k0 + 32);
            stageB(cur ^ 1, k0 + 32);
        }
        // ---- 4 wmma
        #pragma unroll
        for (int t = 0; t < 2; ++t)
            #pragma unroll
            for (int u = 0; u < 2; ++u)
                acc[t][u] = __builtin_amdgcn_wmma_f32_16x16x32_f16(
                    false, af[t], false, bf[u], (short)0, acc[t][u], false, false);

        __syncthreads();
        cur ^= 1;
    }

    // ---- epilogue: NHWC pixel index == m; constexpr strides
    #pragma unroll
    for (int t = 0; t < 2; ++t) {
        const int mBase = m0 + wm * 32 + t * 16 + g * 8;
        #pragma unroll
        for (int u = 0; u < 2; ++u) {
            const int n = nBlock + wn * 32 + u * 16 + hl;
            _Float16* dst = out + (long)mBase * OCT + CHOFF + n;
            #pragma unroll
            for (int r = 0; r < 8; ++r) {
                float v = acc[t][u][r];
                if (RELU) v = v > 0.0f ? v : 0.0f;
                dst[(long)r * OCT] = (_Float16)v;
            }
        }
    }
}

// ---------------------------------------------------------------------------
// Weight transform: w[oc][ic][kh][kw] f32 -> wT[oc][cell][ic] f16, K padded
// to KP with zeros. (kh,kw) flat index == cell, so src = w[(n*IC+ic)*KHW+cell].
// ---------------------------------------------------------------------------
__global__ void weight_transform_kernel(const float* __restrict__ w,
                                        _Float16* __restrict__ wt,
                                        int OC, int IC, int KHW, int K, int KP)
{
    const int t = blockIdx.x * blockDim.x + threadIdx.x;
    if (t >= OC * KP) return;
    const int n  = t / KP;
    const int kp = t - n * KP;
    _Float16 v = (_Float16)0.0f;
    if (kp < K) {
        const int cell = kp / IC;
        const int ic   = kp - cell * IC;
        v = (_Float16)w[(long)(n * IC + ic) * KHW + cell];
    }
    wt[t] = v;
}

// x (B,1,64,64) f32 -> xs NHWC f16 (same order, C=1) and xt NWC f16 (transposed)
__global__ void input_convert_kernel(const float* __restrict__ x,
                                     _Float16* __restrict__ xs,
                                     _Float16* __restrict__ xt)
{
    const int t = blockIdx.x * blockDim.x + threadIdx.x;
    if (t >= BATCH * 4096) return;
    const float v = x[t];
    xs[t] = (_Float16)v;
    const int b = t >> 12, h = (t >> 6) & 63, w = t & 63;
    xt[(b << 12) + (w << 6) + h] = (_Float16)v;
}

// t3 NWC (B,64w,256c) -> concat channels 0..3:  cat[b][h][w][tc] = t3[b][w][tc*64+h]
__global__ void t3_to_cat_kernel(const _Float16* __restrict__ t3,
                                 _Float16* __restrict__ cat)
{
    const int t = blockIdx.x * blockDim.x + threadIdx.x;  // ((b*64+h)*64+w)*4+tc
    if (t >= BATCH * 4096 * 4) return;
    const int tc = t & 3;
    const int pix = t >> 2;
    const int w = pix & 63, h = (pix >> 6) & 63, b = pix >> 12;
    cat[(long)pix * 260 + tc] = t3[((b << 6) + w) * 256 + tc * 64 + h];
}

// ---------------------------------------------------------------------------
// Primary capsule squash: p NHWC f16 (B,784,256) -> u f32 (B, 25088, 8),
// i = cap*784 + pos; the 8 dims are contiguous halves.
// ---------------------------------------------------------------------------
__global__ void primary_squash_kernel(const _Float16* __restrict__ p,
                                      float* __restrict__ u)
{
    const int t = blockIdx.x * blockDim.x + threadIdx.x;  // b*25088 + cap*784 + pos
    if (t >= BATCH * 32 * 784) return;
    const int pos = t % 784;
    const int rem = t / 784;
    const int cap = rem & 31;
    const int b   = rem >> 5;
    const _Float16* src = p + ((long)(b * 784 + pos)) * 256 + cap * 8;
    float vals[8];
    float sn = 0.0f;
    #pragma unroll
    for (int d = 0; d < 8; ++d) {
        const float x = (float)src[d];
        vals[d] = x;
        sn += x * x;
    }
    const float scale = sn / ((1.0f + sn) * sqrtf(sn + 1e-8f));
    const long base = (long)t * 8;
    #pragma unroll
    for (int d = 0; d < 8; ++d) u[base + d] = vals[d] * scale;
}

// ---------------------------------------------------------------------------
// Fused routing pass (u_hat recomputed each iteration; cheaper in bytes than
// materializing it).  mode 0: c=0.5 ; mode 1: b=uhat.v0, store, softmax ;
// mode 2: b=bij+uhat.v1, softmax.
// ---------------------------------------------------------------------------
#define IN_CAPS 25088
__global__ void __launch_bounds__(256) routing_kernel(
    const float* __restrict__ wr, const float* __restrict__ u,
    const float* __restrict__ v, float* __restrict__ bij,
    float* __restrict__ s, int mode)
{
    __shared__ float s_acc[32];
    __shared__ float sv[32];
    const int b = blockIdx.y;
    const int i = blockIdx.x * blockDim.x + threadIdx.x;
    if (threadIdx.x < 32) {
        s_acc[threadIdx.x] = 0.0f;
        sv[threadIdx.x] = (mode != 0) ? v[b * 32 + threadIdx.x] : 0.0f;
    }
    __syncthreads();

    if (i < IN_CAPS) {
        float uv[8];
        #pragma unroll
        for (int k = 0; k < 8; ++k) uv[k] = u[((long)b * IN_CAPS + i) * 8 + k];

        float uhat[32];
        const float* w = wr + (long)i * 256;
        #pragma unroll
        for (int od = 0; od < 32; ++od) {
            float a = 0.0f;
            #pragma unroll
            for (int k = 0; k < 8; ++k) a += w[od * 8 + k] * uv[k];
            uhat[od] = a;
        }

        float c0, c1;
        if (mode == 0) {
            c0 = c1 = 0.5f;
        } else {
            float d0 = 0.0f, d1 = 0.0f;
            #pragma unroll
            for (int d = 0; d < 16; ++d) {
                d0 += uhat[d]      * sv[d];
                d1 += uhat[16 + d] * sv[16 + d];
            }
            float b0, b1;
            const long bidx = ((long)b * IN_CAPS + i) * 2;
            if (mode == 1) {
                b0 = d0; b1 = d1;
                bij[bidx + 0] = b0; bij[bidx + 1] = b1;
            } else {
                b0 = bij[bidx + 0] + d0;
                b1 = bij[bidx + 1] + d1;
            }
            const float mx = fmaxf(b0, b1);
            const float e0 = __expf(b0 - mx), e1 = __expf(b1 - mx);
            const float inv = 1.0f / (e0 + e1);
            c0 = e0 * inv; c1 = e1 * inv;
        }
        #pragma unroll
        for (int d = 0; d < 16; ++d) {
            atomicAdd(&s_acc[d],      c0 * uhat[d]);
            atomicAdd(&s_acc[16 + d], c1 * uhat[16 + d]);
        }
    }
    __syncthreads();
    if (threadIdx.x < 32) atomicAdd(&s[b * 32 + threadIdx.x], s_acc[threadIdx.x]);
}

__global__ void squash_v_kernel(const float* __restrict__ s,
                                float* __restrict__ v, int absout)
{
    const int t = blockIdx.x * blockDim.x + threadIdx.x;
    if (t >= BATCH * 32) return;
    const float x  = s[t];
    const float sn = x * x;
    const float r  = sn * x / ((1.0f + sn) * sqrtf(sn + 1e-8f));
    v[t] = absout ? fabsf(r) : r;
}

__global__ void zero_kernel(float* __restrict__ p, int n)
{
    const int t = blockIdx.x * blockDim.x + threadIdx.x;
    if (t < n) p[t] = 0.0f;
}

// ---------------------------------------------------------------------------
extern "C" void kernel_launch(void* const* d_in, const int* in_sizes, int n_in,
                              void* d_out, int out_size, void* d_ws, size_t ws_size,
                              hipStream_t stream)
{
    const float* x    = (const float*)d_in[0];
    const float* w_t1 = (const float*)d_in[1];  const float* b_t1 = (const float*)d_in[2];
    const float* w_t2 = (const float*)d_in[3];  const float* b_t2 = (const float*)d_in[4];
    const float* w_t3 = (const float*)d_in[5];  const float* b_t3 = (const float*)d_in[6];
    const float* w_s1 = (const float*)d_in[7];  const float* b_s1 = (const float*)d_in[8];
    const float* w_s2 = (const float*)d_in[9];  const float* b_s2 = (const float*)d_in[10];
    const float* w_s3 = (const float*)d_in[11]; const float* b_s3 = (const float*)d_in[12];
    const float* w_f  = (const float*)d_in[13]; const float* b_f  = (const float*)d_in[14];
    const float* w_p  = (const float*)d_in[15]; const float* b_p  = (const float*)d_in[16];
    const float* w_r  = (const float*)d_in[17];
    float* outp = (float*)d_out;

    char* ws = (char*)d_ws;
    size_t off = 0;
    auto allocH = [&](size_t halves) -> _Float16* {
        _Float16* p = (_Float16*)(ws + off);
        off += (halves * 2 + 255) & ~(size_t)255;
        return p;
    };
    auto allocF = [&](size_t floats) -> float* {
        float* p = (float*)(ws + off);
        off += (floats * 4 + 255) & ~(size_t)255;
        return p;
    };
    // swizzled f16 weights (K padded to KP)
    _Float16* wtT1 = allocH(64  * 320);
    _Float16* wtT2 = allocH(128 * 320);
    _Float16* wtT3 = allocH(256 * 640);
    _Float16* wtS1 = allocH(64  * 32);
    _Float16* wtS2 = allocH(128 * 576);
    _Float16* wtS3 = allocH(256 * 1152);
    _Float16* wtF  = allocH(256 * 288);
    _Float16* wtP  = allocH(256 * 20736);
    // activations (NHWC / NWC f16)
    _Float16* xs   = allocH((size_t)BATCH * 4096);         // (B,64,64,1)
    _Float16* xt   = allocH((size_t)BATCH * 4096);         // (B,64w,64c)
    _Float16* t1b  = allocH((size_t)BATCH * 64 * 64);
    _Float16* t2b  = allocH((size_t)BATCH * 64 * 128);
    _Float16* t3b  = allocH((size_t)BATCH * 64 * 256);
    _Float16* sp1  = allocH((size_t)BATCH * 4096 * 64);
    _Float16* sp2  = allocH((size_t)BATCH * 4096 * 128);
    _Float16* catb = allocH((size_t)BATCH * 4096 * 260 + 128); // +guard for k>=K reads
    _Float16* fusd = allocH((size_t)BATCH * 4096 * 256);
    _Float16* pbuf = allocH((size_t)BATCH * 784 * 256);
    float*    ubuf = allocF((size_t)BATCH * IN_CAPS * 8);
    float*    bij  = allocF((size_t)BATCH * IN_CAPS * 2);
    float*    sbuf = allocF(BATCH * 32);
    float*    vbuf = allocF(BATCH * 32);

    // ---- input conversion + weight swizzle
    input_convert_kernel<<<(BATCH * 4096 + 255) / 256, 256, 0, stream>>>(x, xs, xt);
    auto wxf = [&](const float* w, _Float16* wt, int OC, int IC, int KHW, int KP) {
        const int K = IC * KHW;
        weight_transform_kernel<<<(OC * KP + 255) / 256, 256, 0, stream>>>(w, wt, OC, IC, KHW, K, KP);
    };
    wxf(w_t1, wtT1, 64,  64,  5,  320);
    wxf(w_t2, wtT2, 128, 64,  5,  320);
    wxf(w_t3, wtT3, 256, 128, 5,  640);
    wxf(w_s1, wtS1, 64,  1,   9,  32);
    wxf(w_s2, wtS2, 128, 64,  9,  576);
    wxf(w_s3, wtS3, 256, 128, 9,  1152);
    wxf(w_f,  wtF,  256, 260, 1,  288);
    wxf(w_p,  wtP,  256, 256, 81, 20736);

    const dim3 blk(128);
    // grid = (OC/64, BATCH*OH*OW/64)

    // ---- temporal branch (Conv1d as IH=1/KH=1 conv), relu, NWC layout
    wmma_conv_kernel<64, 64, 1, 64, 1, 5, 1, 1, 0, 2, 1, 64, 64, 0, 1>
        <<<dim3(1, 32), blk, 0, stream>>>(xt, wtT1, b_t1, t1b);
    wmma_conv_kernel<128, 64, 1, 64, 1, 5, 1, 1, 0, 2, 1, 64, 128, 0, 1>
        <<<dim3(2, 32), blk, 0, stream>>>(t1b, wtT2, b_t2, t2b);
    wmma_conv_kernel<256, 128, 1, 64, 1, 5, 1, 1, 0, 2, 1, 64, 256, 0, 1>
        <<<dim3(4, 32), blk, 0, stream>>>(t2b, wtT3, b_t3, t3b);
    t3_to_cat_kernel<<<(BATCH * 4096 * 4 + 255) / 256, 256, 0, stream>>>(t3b, catb);

    // ---- spectral branch: 3x3 pad-1 relu; s3 -> concat channels 4..259
    wmma_conv_kernel<64, 1, 64, 64, 3, 3, 1, 1, 1, 1, 64, 64, 64, 0, 1>
        <<<dim3(1, 2048), blk, 0, stream>>>(xs, wtS1, b_s1, sp1);
    wmma_conv_kernel<128, 64, 64, 64, 3, 3, 1, 1, 1, 1, 64, 64, 128, 0, 1>
        <<<dim3(2, 2048), blk, 0, stream>>>(sp1, wtS2, b_s2, sp2);
    wmma_conv_kernel<256, 128, 64, 64, 3, 3, 1, 1, 1, 1, 64, 64, 260, 4, 1>
        <<<dim3(4, 2048), blk, 0, stream>>>(sp2, wtS3, b_s3, catb);

    // ---- fusion 1x1 (no relu), primary caps 9x9 s2 (no relu)
    wmma_conv_kernel<256, 260, 64, 64, 1, 1, 1, 1, 0, 0, 64, 64, 256, 0, 0>
        <<<dim3(4, 2048), blk, 0, stream>>>(catb, wtF, b_f, fusd);
    wmma_conv_kernel<256, 256, 64, 64, 9, 9, 2, 2, 0, 0, 28, 28, 256, 0, 0>
        <<<dim3(4, 392), blk, 0, stream>>>(fusd, wtP, b_p, pbuf);

    // ---- capsule squash: p -> u (B, 25088, 8) f32
    primary_squash_kernel<<<(BATCH * 32 * 784 + 255) / 256, 256, 0, stream>>>(pbuf, ubuf);

    // ---- dynamic routing, 3 iterations
    const dim3 rg((IN_CAPS + 255) / 256, BATCH);
    zero_kernel<<<4, 256, 0, stream>>>(sbuf, BATCH * 32);
    routing_kernel<<<rg, 256, 0, stream>>>(w_r, ubuf, vbuf, bij, sbuf, 0);
    squash_v_kernel<<<4, 256, 0, stream>>>(sbuf, vbuf, 0);

    zero_kernel<<<4, 256, 0, stream>>>(sbuf, BATCH * 32);
    routing_kernel<<<rg, 256, 0, stream>>>(w_r, ubuf, vbuf, bij, sbuf, 1);
    squash_v_kernel<<<4, 256, 0, stream>>>(sbuf, vbuf, 0);

    zero_kernel<<<4, 256, 0, stream>>>(sbuf, BATCH * 32);
    routing_kernel<<<rg, 256, 0, stream>>>(w_r, ubuf, vbuf, bij, sbuf, 2);
    squash_v_kernel<<<4, 256, 0, stream>>>(sbuf, outp, 1);   // |v| -> (B, 2, 16)

    (void)in_sizes; (void)n_in; (void)out_size; (void)ws_size;
}
